// CgpHmmCell_54597624267474
// MI455X (gfx1250) — compile-verified, matrix operand order checked by reading, not run on hardware
//
#include <hip/hip_runtime.h>
#include <stdint.h>

// ---------------- problem constants (from reference) ----------------
#define S_DIM    612          // states
#define SPAD     640          // padded: 40 N-tiles of 16 / 20 K-tiles of 32
#define M_DIM    126          // observation alphabet
#define BATCH    64
#define KT_TILES 20           // SPAD/32

typedef __attribute__((ext_vector_type(16))) _Float16     v16h;
typedef __attribute__((ext_vector_type(8)))  float        v8f;
typedef __attribute__((ext_vector_type(4)))  unsigned int v4u;   // plain 128-bit vector

union Frag { v16h v; v4u q[2]; };

// explicit global-address-space pointer (survives the anti-LICM asm opacity)
typedef const __attribute__((address_space(1))) v4u* gq_ptr;

// ---------------------------------------------------------------------------
// kl (0..31, K within a 32-wide tile) -> (lane-half, half-index) for the CDNA5
// 16-bit WMMA operand layout. Identical mapping for A (16x32) and B (32x16).
// ---------------------------------------------------------------------------
__device__ __forceinline__ void kl_map(int kl, int& hi2, int& h) {
  int quad = kl >> 4;
  int rr   = kl & 7;
  hi2 = (kl >> 3) & 1;
  h   = (quad * 4 + (rr >> 1)) * 2 + (rr & 1);
}

// A-matrix element (k-row, n-col) -> WMMA B-fragment swizzled global storage.
__device__ __forceinline__ void storeA_swizzled(_Float16* Ah, int i, int j, float val) {
  int kt = i >> 5, kl = i & 31;
  int nt = j >> 4, nl = j & 15;
  int hi2, h; kl_map(kl, hi2, h);
  int lane = nl + (hi2 << 4);
  Ah[((((nt * KT_TILES) + kt) * 32 + lane) << 4) + h] = (_Float16)val;
}

// alpha element (m-row, col) -> WMMA A-fragment swizzled LDS storage.
__device__ __forceinline__ void store_alpha(_Float16* aswz, int m, int col, float val) {
  int kt = col >> 5, kl = col & 31;
  int hi2, h; kl_map(kl, hi2, h);
  int lane = m + (hi2 << 4);
  aswz[((kt * 32 + lane) << 4) + h] = (_Float16)val;
}

// ---------------- build kernel 1: clear dense A (f16) and Bm (f32) ----------
__global__ void build_zero(unsigned int* Ah32, float* Bm, int nA32, int nBm) {
  int stride = gridDim.x * blockDim.x;
  for (int k = blockIdx.x * blockDim.x + threadIdx.x; k < nA32; k += stride) Ah32[k] = 0u;
  for (int k = blockIdx.x * blockDim.x + threadIdx.x; k < nBm;  k += stride) Bm[k]  = 0.f;
}

// ---------------- build kernel 2: scatter weights / constants ---------------
__global__ void build_scatter(_Float16* Ah, float* Bm,
                              const float* tw, const int* ia_w, int n_aw,
                              const int* ia_c, int n_ac,
                              const float* ew, const int* ib_w, int n_bw,
                              const int* ib_c, int n_bc) {
  int i = blockIdx.x * blockDim.x + threadIdx.x;
  int total = n_aw + n_ac + n_bw + n_bc;
  if (i >= total) return;
  if (i < n_aw) {
    storeA_swizzled(Ah, ia_w[2 * i], ia_w[2 * i + 1], tw[i]);
  } else if (i < n_aw + n_ac) {
    int k = i - n_aw;
    storeA_swizzled(Ah, ia_c[2 * k], ia_c[2 * k + 1], 1.0f);
  } else if (i < n_aw + n_ac + n_bw) {
    int k = i - n_aw - n_ac;
    Bm[ib_w[2 * k] * SPAD + ib_w[2 * k + 1]] = ew[k];
  } else {
    int k = i - n_aw - n_ac - n_bw;
    Bm[ib_c[2 * k] * SPAD + ib_c[2 * k + 1]] = 1.0f;
  }
}

// ---------------------------------------------------------------------------
// Forward scan. 4 blocks x 256 threads (8 waves). Block bt owns batch rows
// [bt*16, bt*16+16). alpha lives in LDS in WMMA A-fragment layout. Loop nest:
// kt outer / N-tile inner -> each alpha fragment is loaded ONCE per wave per
// step (20 x 2 ds_load_b128) and feeds 5 WMMAs into 5 independent
// accumulators; A-matrix streams from L2 as pre-swizzled B-fragments
// (global_load_b128 via explicit addrspace(1) pointer).
// ---------------------------------------------------------------------------
__global__ __launch_bounds__(256) void hmm_scan(const _Float16* __restrict__ Ah,
                                                const float* __restrict__ Bm,
                                                const float* __restrict__ I,
                                                const int* __restrict__ obs,
                                                float* __restrict__ ll_out,
                                                int T) {
  __shared__ _Float16 aswz[KT_TILES * 32 * 16];   // 20 KB, A-fragment layout
  __shared__ float zsum2[2][16];
  __shared__ float ll_s[16];
  __shared__ int   o_s2[2][16];

  const int tid  = threadIdx.x;
  const int lane = tid & 31;
  const int wid  = tid >> 5;
  const int ml   = lane & 15;     // N within tile (B/C/D layout)
  const int hi   = lane >> 4;     // lane-half selector
  const int bt   = blockIdx.x;

  // ---- t = 0: alpha0 = I * Bm[obs[:,0]], normalize, ll0 = log Z0 ----
  for (int idx = tid; idx < KT_TILES * 32 * 16; idx += 256) aswz[idx] = (_Float16)0.f;
  if (tid < 16) {
    zsum2[0][tid] = 0.f;
    zsum2[1][tid] = 0.f;           // preset for first scan step (parity 1)
    ll_s[tid]     = 0.f;
    o_s2[0][tid]  = obs[(bt * 16 + tid) * T + 0];
    o_s2[1][tid]  = obs[(bt * 16 + tid) * T + 1];
  }
  __syncthreads();
  for (int e = tid; e < 16 * S_DIM; e += 256) {
    int m = e / S_DIM, col = e - m * S_DIM;
    atomicAdd(&zsum2[0][m], I[col] * Bm[o_s2[0][m] * SPAD + col]);
  }
  __syncthreads();
  if (tid < 16) ll_s[tid] = __logf(zsum2[0][tid]);
  for (int e = tid; e < 16 * S_DIM; e += 256) {
    int m = e / S_DIM, col = e - m * S_DIM;
    float a0 = I[col] * Bm[o_s2[0][m] * SPAD + col];
    store_alpha(aswz, m, col, a0 / zsum2[0][m]);
  }

  // ---- sequential scan over t = 1..T-1 (2 barriers per step) ----
  for (int t = 1; t < T; ++t) {
    const int p = t & 1;
    __syncthreads();               // prev normalize stores + preps visible

    // Opaque base (per step) stops LICM hoisting the 100 B-fragments out of
    // the t-loop; addrspace(1) cast keeps the loads as global_load_b128.
    uintptr_t ap = (uintptr_t)Ah;
    asm volatile("" : "+s"(ap));
    // v4u-granular base index for this wave's first tile at kt=0:
    // halves offset = ((nt0*KT + kt)*32 + lane)*16 ; 8 halves per v4u
    gq_ptr gb = (gq_ptr)ap;
    const int base_q = (((wid * 5) * KT_TILES) * 32 + lane) * 2;

    int om[8];
    #pragma unroll
    for (int r = 0; r < 8; ++r) om[r] = o_s2[p][r + 8 * hi];

    // ---- GEMM: kt outer, alpha fragment loaded once, 5 accumulators ----
    v8f acc[5];
    #pragma unroll
    for (int i = 0; i < 5; ++i) acc[i] = (v8f){};
    #pragma unroll
    for (int kt = 0; kt < KT_TILES; ++kt) {
      Frag a;
      const v4u* pA = (const v4u*)&aswz[(kt * 32 + lane) << 4];
      a.q[0] = pA[0]; a.q[1] = pA[1];
      #pragma unroll
      for (int i = 0; i < 5; ++i) {
        Frag b;
        const int q = base_q + (i * KT_TILES + kt) * 64;  // 32 lanes * 2 v4u
        b.q[0] = gb[q];
        b.q[1] = gb[q + 1];
        acc[i] = __builtin_amdgcn_wmma_f32_16x16x32_f16(
            false, a.v, false, b.v, (short)0, acc[i], false, false);
      }
    }

    // ---- e-multiply in D-layout, partial row-sums for Z ----
    float v[5][8];
    #pragma unroll
    for (int i = 0; i < 5; ++i) {
      const int col = (wid * 5 + i) * 16 + ml;
      #pragma unroll
      for (int r = 0; r < 8; ++r) {
        float val = 0.f;
        if (col < S_DIM) val = acc[i][r] * Bm[om[r] * SPAD + col];
        v[i][r] = val;
        float red = val;
        red += __shfl_xor(red, 1, 32);
        red += __shfl_xor(red, 2, 32);
        red += __shfl_xor(red, 4, 32);
        red += __shfl_xor(red, 8, 32);
        if (ml == 0) atomicAdd(&zsum2[p][r + 8 * hi], red);  // ds_add_f32
      }
    }
    __syncthreads();               // Z complete

    if (tid < 16) ll_s[tid] += __logf(zsum2[p][tid]);
    float zinv[8];
    #pragma unroll
    for (int r = 0; r < 8; ++r) zinv[r] = 1.0f / zsum2[p][r + 8 * hi];

    // ---- normalize, convert to f16, scatter into swizzled alpha ----
    #pragma unroll
    for (int i = 0; i < 5; ++i) {
      const int col = (wid * 5 + i) * 16 + ml;
      if (col < S_DIM) {
        // element (m = r+8*hi, col): lane2 = m + 16*hi2 -> addr separates as
        // base + (r<<4); compute base once per tile.
        int kt = col >> 5, kl = col & 31;
        int hi2, h; kl_map(kl, hi2, h);
        _Float16* basep = &aswz[((kt * 32 + 8 * hi + (hi2 << 4)) << 4) + h];
        #pragma unroll
        for (int r = 0; r < 8; ++r)
          basep[r << 4] = (_Float16)(v[i][r] * zinv[r]);
      }
    }
    // prep next step's buffers (other parity; read only after next barrier)
    if (tid < 16) {
      zsum2[p ^ 1][tid] = 0.f;
      if (t + 1 < T) o_s2[p ^ 1][tid] = obs[(bt * 16 + tid) * T + t + 1];
    }
  }

  __syncthreads();
  if (tid < 16) ll_out[bt * 16 + tid] = ll_s[tid];
}

// ---------------------------------------------------------------------------
extern "C" void kernel_launch(void* const* d_in, const int* in_sizes, int n_in,
                              void* d_out, int out_size, void* d_ws, size_t ws_size,
                              hipStream_t stream) {
  const float* init_kernel       = (const float*)d_in[0];
  const float* transition_kernel = (const float*)d_in[1];
  const float* emission_kernel   = (const float*)d_in[2];
  const int*   obs               = (const int*)d_in[3];
  const int*   ia_w              = (const int*)d_in[4];
  const int*   ia_c              = (const int*)d_in[5];
  const int*   ib_w              = (const int*)d_in[6];
  const int*   ib_c              = (const int*)d_in[7];
  // d_in[8] = ii == arange(S) -> I is init_kernel densely.

  const int n_aw = in_sizes[1];
  const int n_ac = in_sizes[5] / 2;
  const int n_bw = in_sizes[2];
  const int n_bc = in_sizes[7] / 2;
  const int T    = in_sizes[3] / BATCH;

  // workspace: [Ah: SPAD*SPAD f16 fragment-swizzled][Bm: M_DIM*SPAD f32]
  _Float16* Ah = (_Float16*)d_ws;
  const size_t ah_bytes = (size_t)SPAD * SPAD * sizeof(_Float16);   // 819,200 B
  float* Bm = (float*)((char*)d_ws + ah_bytes);                     // 322,560 B

  const int nA32 = (SPAD * SPAD) / 2;
  const int nBm  = M_DIM * SPAD;
  build_zero<<<512, 256, 0, stream>>>((unsigned int*)Ah, Bm, nA32, nBm);

  const int total = n_aw + n_ac + n_bw + n_bc;
  build_scatter<<<(total + 255) / 256, 256, 0, stream>>>(
      Ah, Bm, transition_kernel, ia_w, n_aw, ia_c, n_ac,
      emission_kernel, ib_w, n_bw, ib_c, n_bc);

  hmm_scan<<<BATCH / 16, 256, 0, stream>>>(Ah, Bm, init_kernel, obs,
                                           (float*)d_out, T);
}
